// ALayer_v1_14637248545081
// MI455X (gfx1250) — matrix operand order repaired and checked by reading, not compiled, not run on hardware
//
#include <hip/hip_runtime.h>
#include <hip/hip_bf16.h>

// ---------------------------------------------------------------------------
// Problem constants (reference: B=32, C=256, H=W=56, mid=16)
// ---------------------------------------------------------------------------
#define BATCH 32
#define CIN   256
#define HH    56
#define WW    56
#define HW    (HH * WW)            // 3136
#define OC1   16
#define NPIX  (BATCH * HW)         // 100352 total pixels

typedef __attribute__((ext_vector_type(16))) __bf16 v16bf;
typedef __attribute__((ext_vector_type(8)))  __bf16 v8bf;
typedef __attribute__((ext_vector_type(8)))  float  v8f;

// ---------------------------------------------------------------------------
// Kernel 0a: repack conv1 weights into WMMA B-fragment order (bf16), and
// zero the 512B padding buffer used by out-of-bounds conv taps.
// B matrix layout (16-bit, 32x16, wave32):
//   lane 0-15  -> N = lane,     elements e=0..15 -> K = (e<8 ? e : e+8)
//   lane 16-31 -> N = lane-16,  elements e=0..15 -> K = 8 + (e<8 ? e : e+8)
// Packed index = (((tap*8 + kb)*32 + lane)*16 + e) so kernel 1 does ONE
// contiguous 32B load per lane per fragment.
// ---------------------------------------------------------------------------
__global__ __launch_bounds__(256) void k_pack_w1(const float* __restrict__ w1,
                                                 __bf16* __restrict__ packB,
                                                 __bf16* __restrict__ zpad) {
    int idx = blockIdx.x * 256 + threadIdx.x;           // 0 .. 9*8*32*16-1
    if (idx < 256) zpad[idx] = (__bf16)0.0f;            // 512B zero pad
    if (idx >= 9 * 8 * 32 * 16) return;
    int e    =  idx        & 15;
    int lane = (idx >> 4)  & 31;
    int kb   = (idx >> 9)  & 7;
    int tap  =  idx >> 12;                              // 0..8
    int n      = lane & 15;                             // output channel
    int klocal = (e < 8 ? e : e + 8) + ((lane >> 4) << 3); // 0..31
    int c      = kb * 32 + klocal;                      // input channel
    int dy = tap / 3, dx = tap % 3;
    float v = w1[((n * CIN + c) * 3 + dy) * 3 + dx];
    packB[idx] = (__bf16)v;
}

// ---------------------------------------------------------------------------
// Kernel 0b: x_in NCHW f32 -> NHWC bf16 via LDS transpose (coalesced both ways)
// One block handles (b, y, 16-pixel x-tile) x all 256 channels.
// ---------------------------------------------------------------------------
__global__ __launch_bounds__(256) void k_nchw_to_nhwc(const float* __restrict__ xin,
                                                      __bf16* __restrict__ xT) {
    int blk = blockIdx.x;                 // b*HH*4 + y*4 + xt
    int xt  = blk & 3;
    int y   = (blk >> 2) % HH;
    int b   = blk / (4 * HH);
    int x0  = xt * 16;
    int npx = (WW - x0) < 16 ? (WW - x0) : 16;          // 16,16,16,8

    __shared__ float tile[CIN * 16];                    // 16 KB
    int t = threadIdx.x;
    for (int i = t; i < CIN * 16; i += 256) {
        int c = i >> 4, px = i & 15;
        float v = 0.f;
        if (px < npx) v = xin[((b * CIN + c) * HH + y) * WW + x0 + px];
        tile[i] = v;
    }
    __syncthreads();
    for (int j = t; j < 16 * CIN; j += 256) {
        int px = j >> 8, c = j & 255;
        if (px < npx)
            xT[(size_t)(((b * HH + y) * WW + x0 + px)) * CIN + c] =
                (__bf16)tile[(c << 4) + px];
    }
}

// ---------------------------------------------------------------------------
// Kernel 1: conv1 (3x3, 256->16) + ReLU as implicit GEMM with bf16 WMMA.
// One wave = one 16-pixel(M) x 16-channel(N) output tile, K = 2304 as
// 9 taps x 8 blocks of 32 -> 72 v_wmma_f32_16x16x32_bf16, fp32 accumulate
// into two interleaved accumulator chains.
// A layout (16-bit, 16x32): lane -> M = lane&15;
//   elements e: K = (e<8 ? e : e+8) + (lane>=16 ? 8 : 0)
//   => lanes 0-15 need c-chunks [0..7] and [16..23]; lanes 16-31 [8..15],[24..31]
//   => two contiguous 16B (8 x bf16) global loads per lane (NHWC layout).
// Out-of-bounds taps read a zeroed 512B pad buffer instead of branching, so
// EXEC stays all-1s through the whole main loop (no saveexec, no zero-fills).
// ---------------------------------------------------------------------------
__global__ __launch_bounds__(128) void k_conv1_wmma(const __bf16* __restrict__ xT,
                                                    const __bf16* __restrict__ packB,
                                                    const __bf16* __restrict__ zpad,
                                                    __bf16* __restrict__ h) {
    int wave = (blockIdx.x << 2) + (threadIdx.x >> 5);  // tile id, 0..6271
    int lane = threadIdx.x & 31;
    int m    = lane & 15;
    int sel8 = (lane >> 4) << 3;                        // 0 or 8 (c-chunk select)

    int p   = wave * 16 + m;                            // this lane's pixel
    int b   = p / HW;
    int rem = p - b * HW;
    int y   = rem / WW;
    int x   = rem - y * WW;

    v8f acc0 = {};
    v8f acc1 = {};
    for (int tap = 0; tap < 9; ++tap) {
        int dy = tap / 3, dx = tap % 3;
        int yy = y + dy - 1, xx = x + dx - 1;
        bool valid = (yy >= 0) && (yy < HH) && (xx >= 0) && (xx < WW);
        // invalid lanes read zeros from the pad buffer: no EXEC divergence
        const __bf16* ap =
            valid ? (xT + (size_t)((b * HH + yy) * WW + xx) * CIN) : zpad;
#pragma unroll
        for (int kb = 0; kb < 8; ++kb) {
            v8bf lo = *(const v8bf*)(ap + kb * 32 + sel8);       // K [c0..c0+7]
            v8bf hi = *(const v8bf*)(ap + kb * 32 + sel8 + 16);  // K [c0+16..c0+23]
            v16bf afrag;
#pragma unroll
            for (int i = 0; i < 8; ++i) { afrag[i] = lo[i]; afrag[8 + i] = hi[i]; }
            v16bf bfrag = *(const v16bf*)(packB + ((((tap * 8 + kb) * 32) + lane) << 4));
            if (kb & 1)
                acc1 = __builtin_amdgcn_wmma_f32_16x16x32_bf16(
                    false, afrag, false, bfrag, (short)0, acc1, false, false);
            else
                acc0 = __builtin_amdgcn_wmma_f32_16x16x32_bf16(
                    false, afrag, false, bfrag, (short)0, acc0, false, false);
        }
    }

    // C/D layout: lane 0-15 -> N=lane, M=v; lane 16-31 -> N=lane-16, M=v+8
    int n  = lane & 15;
    int mb = sel8;                                      // row base 0 or 8
#pragma unroll
    for (int v = 0; v < 8; ++v) {
        int pp = wave * 16 + mb + v;
        int bb = pp / HW;
        int r2 = pp - bb * HW;
        float val = acc0[v] + acc1[v];
        val = val > 0.f ? val : 0.f;                    // fused ReLU
        h[(size_t)(bb * OC1 + n) * HW + r2] = (__bf16)val;
    }
}

// ---------------------------------------------------------------------------
// Kernel 2: conv2 (3x3, 16->1) + sigmoid -> A[b,y,x]. Tiny; VALU.
// ---------------------------------------------------------------------------
__global__ __launch_bounds__(256) void k_conv2_sigmoid(const __bf16* __restrict__ h,
                                                       const float* __restrict__ w2,
                                                       float* __restrict__ A) {
    __shared__ float w2s[OC1 * 9];
    int t = threadIdx.x;
    if (t < OC1 * 9) w2s[t] = w2[t];
    __syncthreads();

    int p = blockIdx.x * 256 + t;                       // 392*256 == NPIX exactly
    int b = p / HW;
    int rem = p - b * HW;
    int y = rem / WW;
    int x = rem - y * WW;

    float s = 0.f;
#pragma unroll
    for (int ic = 0; ic < OC1; ++ic) {
        const __bf16* hb = h + (size_t)(b * OC1 + ic) * HW;
#pragma unroll
        for (int dy = 0; dy < 3; ++dy) {
            int yy = y + dy - 1;
            if (yy < 0 || yy >= HH) continue;
#pragma unroll
            for (int dx = 0; dx < 3; ++dx) {
                int xx = x + dx - 1;
                if (xx < 0 || xx >= WW) continue;
                s += w2s[(ic * 3 + dy) * 3 + dx] * (float)hb[yy * WW + xx];
            }
        }
    }
    A[p] = 1.f / (1.f + __expf(-s));
}

// ---------------------------------------------------------------------------
// Kernel 3: out = x_out * boxsum3x3(A), broadcast over 256 channels.
// One block per (b, y): box-sum row cached in LDS, then a pure float4
// (global_load_b128 / global_store_b128) stream over 256 channels x 56 px.
// This is the bandwidth-critical pass (~206 MB).
// ---------------------------------------------------------------------------
__global__ __launch_bounds__(256) void k_modulate(const float* __restrict__ A,
                                                  const float* __restrict__ xout,
                                                  float* __restrict__ out) {
    int blk = blockIdx.x;                               // b*HH + y
    int y = blk % HH;
    int b = blk / HH;

    __shared__ float abox[WW];
    int t = threadIdx.x;
    if (t < WW) {
        float s = 0.f;
#pragma unroll
        for (int dy = -1; dy <= 1; ++dy) {
            int yy = y + dy;
            if (yy < 0 || yy >= HH) continue;
            const float* ar = A + (size_t)b * HW + yy * WW;
#pragma unroll
            for (int dx = -1; dx <= 1; ++dx) {
                int xx = t + dx;
                if (xx < 0 || xx >= WW) continue;
                s += ar[xx];
            }
        }
        abox[t] = s;
    }
    __syncthreads();

    // 256 channels * 14 float4 per row = 3584 float4 per block
    const float4* xi = (const float4*)xout;
    float4*       xo = (float4*)out;
    for (int j = t; j < CIN * 14; j += 256) {
        int c  = j / 14;
        int xq = j - c * 14;
        size_t idx4 = (size_t)(b * CIN + c) * (HW / 4) + y * (WW / 4) + xq;
        float4 v = xi[idx4];
        int x0 = xq * 4;
        v.x *= abox[x0 + 0];
        v.y *= abox[x0 + 1];
        v.z *= abox[x0 + 2];
        v.w *= abox[x0 + 3];
        xo[idx4] = v;
    }
}

// ---------------------------------------------------------------------------
// Host-side launcher
// ---------------------------------------------------------------------------
extern "C" void kernel_launch(void* const* d_in, const int* in_sizes, int n_in,
                              void* d_out, int out_size, void* d_ws, size_t ws_size,
                              hipStream_t stream) {
    const float* x_in  = (const float*)d_in[0];
    const float* x_out = (const float*)d_in[1];
    const float* w1    = (const float*)d_in[2];
    const float* w2    = (const float*)d_in[3];
    float* out = (float*)d_out;

    // Workspace layout (all offsets 256B aligned):
    //   xT    : NPIX*CIN bf16      = 51,380,224 B
    //   packB : 36864 bf16         =     73,728 B
    //   h     : BATCH*OC1*HW bf16  =  3,211,264 B
    //   A     : NPIX f32           =    401,408 B
    //   zpad  : 512 B zeros (OOB conv-tap reads)
    char* ws = (char*)d_ws;
    __bf16* xT    = (__bf16*)(ws);
    __bf16* packB = (__bf16*)(ws + 51380224);
    __bf16* h     = (__bf16*)(ws + 51380224 + 73728);
    float*  A     = (float*) (ws + 51380224 + 73728 + 3211264);
    __bf16* zpad  = (__bf16*)(ws + 51380224 + 73728 + 3211264 + 401408);

    k_pack_w1      <<<144, 256, 0, stream>>>(w1, packB, zpad);
    k_nchw_to_nhwc <<<BATCH * HH * 4, 256, 0, stream>>>(x_in, xT);
    k_conv1_wmma   <<<(NPIX / 16) / 4, 128, 0, stream>>>(xT, packB, zpad, h);
    k_conv2_sigmoid<<<NPIX / 256, 256, 0, stream>>>(h, w2, A);
    k_modulate     <<<BATCH * HH, 256, 0, stream>>>(A, x_out, out);
}
